// TransformerLM_79620103733528
// MI455X (gfx1250) — matrix operand order repaired and checked by reading, not compiled, not run on hardware
//
#include <hip/hip_runtime.h>
#include <hip/hip_bf16.h>
#include <stdint.h>

// ---------------------------------------------------------------------------
// Transformer-XL layer for MI455X (gfx1250, wave32, WMMA).
// All dense contractions use v_wmma_f32_16x16x32_f16 with fp32 accumulation.
// All WMMA fragments are contiguous per-lane vector loads (no scalar gathers).
// ---------------------------------------------------------------------------

typedef _Float16 h16_t;
typedef __attribute__((ext_vector_type(16))) _Float16 v16h;
typedef __attribute__((ext_vector_type(8)))  _Float16 v8h;
typedef __attribute__((ext_vector_type(8)))  float    v8f;

#define XLEN   1024
#define MEMLEN 1024
#define TOT    2048
#define BATCH  4
#define DMODEL 1024
#define NHEAD  16
#define DHEAD  64
#define DFF    4096
#define LNEPS  1e-5f

static __device__ __forceinline__ v8f zero_v8f() {
    v8f z = {0.f, 0.f, 0.f, 0.f, 0.f, 0.f, 0.f, 0.f};
    return z;
}

// A-fragment (16x32 f16, ISA 7.12.2): lane holds K = {8h..8h+7} then {16+8h..16+8h+7}
// of its row -> two aligned 16-B chunks.
static __device__ __forceinline__ v16h frag_a(const h16_t* rowp, int half) {
    v8h lo = *(const v8h*)(rowp + 8 * half);
    v8h hi = *(const v8h*)(rowp + 16 + 8 * half);
    v16h r;
#pragma unroll
    for (int j = 0; j < 8; ++j) { r[j] = lo[j]; r[8 + j] = hi[j]; }
    return r;
}
// B-fragment (32x16 f16): lane holds K = {8h..8h+7} then {16+8h..16+8h+7}
// of its column -> two aligned 16-B chunks (same K order as frag_a).
static __device__ __forceinline__ v16h frag_b(const h16_t* colp, int half) {
    v8h lo = *(const v8h*)(colp + 8 * half);
    v8h hi = *(const v8h*)(colp + 16 + 8 * half);
    v16h r;
#pragma unroll
    for (int j = 0; j < 8; ++j) { r[j] = lo[j]; r[8 + j] = hi[j]; }
    return r;
}

// ------------------------------ elementwise --------------------------------

__global__ void k_zero_f32(float* __restrict__ p, int n) {
    int i = blockIdx.x * blockDim.x + threadIdx.x;
    int s = gridDim.x * blockDim.x;
    for (; i < n; i += s) p[i] = 0.f;
}

__global__ void k_cvt_f32_f16(const float* __restrict__ src, h16_t* __restrict__ dst, int n) {
    int i = blockIdx.x * blockDim.x + threadIdx.x;
    int s = gridDim.x * blockDim.x;
    for (; i < n; i += s) dst[i] = (h16_t)src[i];
}

// qu16/qv16 = fp16(q + pos_bias_{u,v});  col = (n*64+d) = i % 1024
__global__ void k_quqv(const float* __restrict__ q, const float* __restrict__ u,
                       const float* __restrict__ v, h16_t* __restrict__ qu,
                       h16_t* __restrict__ qv, int n) {
    int i = blockIdx.x * blockDim.x + threadIdx.x;
    int s = gridDim.x * blockDim.x;
    for (; i < n; i += s) {
        int col = i & (DMODEL - 1);
        float qq = q[i];
        qu[i] = (h16_t)(qq + u[col]);
        qv[i] = (h16_t)(qq + v[col]);
    }
}

// k16 [T*B,1024] fp16 from kv [T*B,2048] (first half)
__global__ void k_splitk(const float* __restrict__ kv, h16_t* __restrict__ k, int n) {
    int i = blockIdx.x * blockDim.x + threadIdx.x;
    int s = gridDim.x * blockDim.x;
    for (; i < n; i += s) {
        int row = i >> 10;
        int col = i & (DMODEL - 1);
        k[i] = (h16_t)kv[(size_t)row * (2 * DMODEL) + col];
    }
}

// VT[b,n,d,j] fp16 from kv[(j*B+b)*2048 + 1024 + n*64 + d] via LDS 64x64 tile.
// Grid: (TOT/64, BATCH*NHEAD), 256 threads. Coalesced on both sides.
__global__ __launch_bounds__(256) void k_transposeV(const float* __restrict__ KV,
                                                    h16_t* __restrict__ VT) {
    __shared__ h16_t t[64][65];
    const int bn = blockIdx.y;            // b*16 + n
    const int b  = bn >> 4, n = bn & 15;
    const int j0 = blockIdx.x * 64;
    const int tid = threadIdx.x;

    const int lj  = tid >> 2;             // 0..63 (j within tile)
    const int ld0 = (tid & 3) * 16;       // 0,16,32,48 (d chunk)
    const float* src = KV + ((size_t)(j0 + lj) * BATCH + b) * (2 * DMODEL)
                        + DMODEL + n * DHEAD + ld0;
#pragma unroll
    for (int e = 0; e < 16; ++e) t[lj][ld0 + e] = (h16_t)src[e];
    __syncthreads();

    const int ldd = tid >> 2;             // d within tile
    const int lj0 = (tid & 3) * 16;       // j chunk
    h16_t* dst = VT + ((size_t)bn * DHEAD + ldd) * TOT + j0 + lj0;
#pragma unroll
    for (int e = 0; e < 16; ++e) dst[e] = t[lj0 + e][ldd];
}

// ------------------------------- layernorm ---------------------------------

__global__ __launch_bounds__(256) void k_layernorm(const float* __restrict__ A,
                                                   const float* __restrict__ R,
                                                   const float* __restrict__ g,
                                                   const float* __restrict__ be,
                                                   float* __restrict__ out) {
    const int row = blockIdx.x;
    const int tid = threadIdx.x;
    const float* a = A + (size_t)row * DMODEL;
    const float* r = R + (size_t)row * DMODEL;
    __shared__ float red[256];

    float s = 0.f;
    for (int c = tid; c < DMODEL; c += 256) s += a[c] + r[c];
    red[tid] = s;
    __syncthreads();
    for (int st = 128; st > 0; st >>= 1) {
        if (tid < st) red[tid] += red[tid + st];
        __syncthreads();
    }
    float mu = red[0] * (1.0f / DMODEL);
    __syncthreads();

    float vv = 0.f;
    for (int c = tid; c < DMODEL; c += 256) {
        float d = a[c] + r[c] - mu;
        vv += d * d;
    }
    red[tid] = vv;
    __syncthreads();
    for (int st = 128; st > 0; st >>= 1) {
        if (tid < st) red[tid] += red[tid + st];
        __syncthreads();
    }
    float rstd = rsqrtf(red[0] * (1.0f / DMODEL) + LNEPS);

    for (int c = tid; c < DMODEL; c += 256)
        out[(size_t)row * DMODEL + c] = (a[c] + r[c] - mu) * rstd * g[c] + be[c];
}

// ------------------------------ WMMA GEMM ----------------------------------
// C[M,N] = A[M,K] * B[K,N] (+bias, +relu). fp16 in, fp32 out.
// Block 256 thr = 8 waves; tile 128x128x32; wave tile 32x64 (2x4 WMMA accs).
// B tile stored TRANSPOSED in LDS (stride 40 halves -> aligned, conflict-free
// b128 fragment loads: bank = 20*lane mod 64 covers {0,4,...,60}).
__global__ __launch_bounds__(256) void k_gemm_f16(const h16_t* __restrict__ A,
                                                  const h16_t* __restrict__ B,
                                                  float* __restrict__ C,
                                                  const float* __restrict__ bias,
                                                  int M, int N, int K, int relu) {
    __shared__ __align__(16) h16_t As[128][32];
    __shared__ __align__(16) h16_t Bs[128][40];   // [col][k], padded stride

    const int tid  = threadIdx.x;
    const int wid  = tid >> 5;
    const int lane = tid & 31;
    const int half = lane >> 4;
    const int rl   = lane & 15;
    const int wm   = wid >> 1;   // 0..3
    const int wn   = wid & 1;    // 0..1
    const int bm   = blockIdx.y * 128;
    const int bn   = blockIdx.x * 128;
    const int arow = tid >> 1, acol = (tid & 1) * 16;
    const int brow = tid >> 3, bcol = (tid & 7) * 16;

    v8f acc[2][4];
#pragma unroll
    for (int i = 0; i < 2; ++i)
#pragma unroll
        for (int j = 0; j < 4; ++j) acc[i][j] = zero_v8f();

    const int ksteps = K >> 5;
    for (int kt = 0; kt < ksteps; ++kt) {
        const h16_t* ga = A + (size_t)(bm + arow) * K + kt * 32 + acol;
        const h16_t* gb = B + (size_t)(kt * 32 + brow) * N + bn + bcol;
        *(v8h*)&As[arow][acol]     = *(const v8h*)ga;
        *(v8h*)&As[arow][acol + 8] = *(const v8h*)(ga + 8);
        v8h b0 = *(const v8h*)gb;
        v8h b1 = *(const v8h*)(gb + 8);
#pragma unroll
        for (int e = 0; e < 8; ++e) {
            Bs[bcol + e][brow]     = b0[e];
            Bs[bcol + 8 + e][brow] = b1[e];
        }
        if (kt + 1 < ksteps) {
            __builtin_prefetch(ga + 32, 0, 1);
            __builtin_prefetch(gb + (size_t)32 * N, 0, 1);
        }
        __syncthreads();

        v16h af[2], bf[4];
#pragma unroll
        for (int sm = 0; sm < 2; ++sm)
            af[sm] = frag_a(&As[wm * 32 + sm * 16 + rl][0], half);
#pragma unroll
        for (int sn = 0; sn < 4; ++sn)
            bf[sn] = frag_b(&Bs[wn * 64 + sn * 16 + rl][0], half);
#pragma unroll
        for (int sm = 0; sm < 2; ++sm)
#pragma unroll
            for (int sn = 0; sn < 4; ++sn)
                acc[sm][sn] = __builtin_amdgcn_wmma_f32_16x16x32_f16(
                    false, af[sm], false, bf[sn], (short)0, acc[sm][sn], false, false);
        __syncthreads();
    }

#pragma unroll
    for (int sm = 0; sm < 2; ++sm)
#pragma unroll
        for (int sn = 0; sn < 4; ++sn)
#pragma unroll
            for (int i2 = 0; i2 < 8; ++i2) {
                int grow = bm + wm * 32 + sm * 16 + i2 + 8 * half;
                int gcol = bn + wn * 64 + sn * 16 + rl;
                float vv = acc[sm][sn][i2];
                if (bias) vv += bias[gcol];
                if (relu) vv = fmaxf(vv, 0.f);
                C[(size_t)grow * N + gcol] = vv;
            }
}

// ------------------------------ attention ----------------------------------
// Grid: (XLEN/64, BATCH*NHEAD), block = 128 (4 waves, 16 query rows each).
// Two-pass softmax; scores = (qu.k^T + shift(qv.r^T)) * scale; rel-shift
// resolved analytically: BD[i,j] = qv_i . r_{j + XLEN-1-i} (unmasked region).
__global__ __launch_bounds__(128) void k_attention(const h16_t* __restrict__ QU,
                                                   const h16_t* __restrict__ QV,
                                                   const h16_t* __restrict__ Kh,
                                                   const h16_t* __restrict__ VT,
                                                   const h16_t* __restrict__ Rh,
                                                   float* __restrict__ vecOut,
                                                   float* __restrict__ attnM) {
    __shared__ __align__(16) h16_t pT[4][16][32];   // prob staging, per wave

    const int wv   = threadIdx.x >> 5;
    const int lane = threadIdx.x & 31;
    const int half = lane >> 4;
    const int rl   = lane & 15;
    const int bh   = blockIdx.y;
    const int b    = bh >> 4;           // / NHEAD
    const int n    = bh & 15;
    const int iw0  = blockIdx.x * 64 + wv * 16;
    const size_t boff = (size_t)b * DMODEL + (size_t)n * DHEAD;
    const int rs = BATCH * DMODEL;      // row stride (halves) over i / j
    const float scale = 0.125f;         // 1/sqrt(64)

    // hoist qu / qv A-fragments (two K=32 chunks of D_HEAD=64)
    v16h aqu[2], aqv[2];
#pragma unroll
    for (int s = 0; s < 2; ++s) {
        aqu[s] = frag_a(QU + (size_t)(iw0 + rl) * rs + boff + s * 32, half);
        aqv[s] = frag_a(QV + (size_t)(iw0 + rl) * rs + boff + s * 32, half);
    }

    // 16x16 score tile for keys [j0, j0+16)
    auto scoreTile = [&](int j0, float* sc) {
        v8f aAC = zero_v8f(), aR0 = zero_v8f(), aR1 = zero_v8f();
        const int u0 = j0 + XLEN - 16 - iw0;   // raw-BD band base (>= 0)
#pragma unroll
        for (int s = 0; s < 2; ++s) {
            v16h bk = frag_b(Kh + (size_t)(j0 + rl) * rs + boff + s * 32, half);
            aAC = __builtin_amdgcn_wmma_f32_16x16x32_f16(false, aqu[s], false, bk,
                                                         (short)0, aAC, false, false);
#pragma unroll
            for (int bnd = 0; bnd < 2; ++bnd) {
                int ur = u0 + bnd * 16 + rl;
                if (ur > TOT - 1) ur = TOT - 1;   // masked region anyway
                v16h br = frag_b(Rh + (size_t)ur * rs + boff + s * 32, half);
                if (bnd == 0)
                    aR0 = __builtin_amdgcn_wmma_f32_16x16x32_f16(false, aqv[s], false, br,
                                                                 (short)0, aR0, false, false);
                else
                    aR1 = __builtin_amdgcn_wmma_f32_16x16x32_f16(false, aqv[s], false, br,
                                                                 (short)0, aR1, false, false);
            }
        }
        // gather the per-row diagonal shift: BD(p,t) = raw(p, t + 15 - p)
#pragma unroll
        for (int i2 = 0; i2 < 8; ++i2) {
            int p = i2 + 8 * half;
            int c = rl + 15 - p;                       // 0..30
            int srcl = (lane & 16) | (c & 15);         // same half, source col
            float bd0 = __shfl(aR0[i2], srcl, 32);
            float bd1 = __shfl(aR1[i2], srcl, 32);
            float bd  = (c < 16) ? bd0 : bd1;
            float val = (aAC[i2] + bd) * scale;
            if ((j0 + rl) > (iw0 + p + MEMLEN)) val = -1e30f;   // causal+mem mask
            sc[i2] = val;
        }
    };

    // causal bound for this wave, rounded up to 32 keys
    int jmax = iw0 + 16 + MEMLEN;
    if (jmax > TOT) jmax = TOT;
    jmax = (jmax + 31) & ~31;
    if (jmax > TOT) jmax = TOT;

    float m[8], den[8];
#pragma unroll
    for (int i2 = 0; i2 < 8; ++i2) { m[i2] = -1e30f; den[i2] = 0.f; }

    // ---- pass 1: row max + denominator (online) ----
    for (int j0 = 0; j0 < jmax; j0 += 16) {
        float sc[8];
        scoreTile(j0, sc);
#pragma unroll
        for (int i2 = 0; i2 < 8; ++i2) {
            float mt = sc[i2];
            mt = fmaxf(mt, __shfl_xor(mt, 1, 32));
            mt = fmaxf(mt, __shfl_xor(mt, 2, 32));
            mt = fmaxf(mt, __shfl_xor(mt, 4, 32));
            mt = fmaxf(mt, __shfl_xor(mt, 8, 32));
            float nm = fmaxf(m[i2], mt);
            float e = __expf(sc[i2] - nm);
            e += __shfl_xor(e, 1, 32);
            e += __shfl_xor(e, 2, 32);
            e += __shfl_xor(e, 4, 32);
            e += __shfl_xor(e, 8, 32);
            den[i2] = den[i2] * __expf(m[i2] - nm) + e;
            m[i2]   = nm;
        }
    }

    float inv[8];
#pragma unroll
    for (int i2 = 0; i2 < 8; ++i2) inv[i2] = 1.0f / den[i2];

    // ---- pass 2: probabilities -> attn_matrix + prob @ V ----
    v8f o[4];
#pragma unroll
    for (int d = 0; d < 4; ++d) o[d] = zero_v8f();

    const h16_t* vtBase = VT + (size_t)bh * DHEAD * TOT;

    for (int j0 = 0; j0 < jmax; j0 += 16) {
        float sc[8];
        scoreTile(j0, sc);
        const int phase = (j0 >> 4) & 1;
#pragma unroll
        for (int i2 = 0; i2 < 8; ++i2) {
            float pv = __expf(sc[i2] - m[i2]) * inv[i2];
            int p = i2 + 8 * half;
            pT[wv][p][phase * 16 + rl] = (h16_t)pv;
            atomicAdd(&attnM[(size_t)(iw0 + p) * TOT + (j0 + rl)], pv * (1.0f / 64.0f));
        }
        if (phase == 1) {
            asm volatile("s_wait_dscnt 0" ::: "memory");   // LDS RAW fence (same wave)
            v16h ap = frag_a(&pT[wv][rl][0], half);
            const int jb0 = j0 - 16;
#pragma unroll
            for (int ds = 0; ds < 4; ++ds) {
                // B frag from transposed V: contiguous in j
                v16h bv = frag_b(vtBase + (size_t)(ds * 16 + rl) * TOT + jb0, half);
                o[ds] = __builtin_amdgcn_wmma_f32_16x16x32_f16(false, ap, false, bv,
                                                               (short)0, o[ds], false, false);
            }
        }
    }

    // vec[i, b, n*64 + d]
#pragma unroll
    for (int ds = 0; ds < 4; ++ds)
#pragma unroll
        for (int i2 = 0; i2 < 8; ++i2) {
            int grow = iw0 + i2 + 8 * half;
            int d    = ds * 16 + rl;
            vecOut[((size_t)grow * BATCH + b) * DMODEL + n * DHEAD + d] = o[ds][i2];
        }
}

// ------------------------------- launcher ----------------------------------

extern "C" void kernel_launch(void* const* d_in, const int* in_sizes, int n_in,
                              void* d_out, int out_size, void* d_ws, size_t ws_size,
                              hipStream_t stream) {
    (void)in_sizes; (void)n_in; (void)out_size; (void)ws_size;

    const float* x    = (const float*)d_in[0];
    const float* mem  = (const float*)d_in[1];
    const float* pos  = (const float*)d_in[2];
    const float* pbu  = (const float*)d_in[3];
    const float* pbv  = (const float*)d_in[4];
    /* d_in[5] = mask (computed analytically) */
    const float* Wq   = (const float*)d_in[6];
    const float* Wkv  = (const float*)d_in[7];
    const float* Wrel = (const float*)d_in[8];
    const float* Wo   = (const float*)d_in[9];
    const float* ln1g = (const float*)d_in[10];
    const float* ln1b = (const float*)d_in[11];
    const float* W1   = (const float*)d_in[12];
    const float* b1   = (const float*)d_in[13];
    const float* W2   = (const float*)d_in[14];
    const float* b2   = (const float*)d_in[15];
    const float* ln2g = (const float*)d_in[16];
    const float* ln2b = (const float*)d_in[17];

    const size_t IB = (size_t)XLEN * BATCH;    // 4096 rows
    const size_t TB = (size_t)TOT * BATCH;     // 8192 rows

    char* wp = (char*)d_ws;
    auto alloc = [&](size_t bytes) -> void* {
        void* p = (void*)wp;
        wp += (bytes + 255) & ~(size_t)255;
        return p;
    };

    h16_t* c16    = (h16_t*)alloc(TB * DMODEL * 2);
    h16_t* pos16  = (h16_t*)alloc(TB * DMODEL * 2);
    h16_t* x16    = (h16_t*)alloc(IB * DMODEL * 2);
    h16_t* Wq16   = (h16_t*)alloc((size_t)DMODEL * DMODEL * 2);
    h16_t* Wkv16  = (h16_t*)alloc((size_t)DMODEL * 2 * DMODEL * 2);
    h16_t* Wrel16 = (h16_t*)alloc((size_t)DMODEL * DMODEL * 2);
    h16_t* Wo16   = (h16_t*)alloc((size_t)DMODEL * DMODEL * 2);
    h16_t* W116   = (h16_t*)alloc((size_t)DMODEL * DFF * 2);
    h16_t* W216   = (h16_t*)alloc((size_t)DFF * DMODEL * 2);
    float* kvf    = (float*)alloc(TB * 2 * DMODEL * 4);
    float* qf     = (float*)alloc(IB * DMODEL * 4);
    float* rf     = (float*)alloc(TB * DMODEL * 4);
    h16_t* qu16   = (h16_t*)alloc(IB * DMODEL * 2);
    h16_t* qv16   = (h16_t*)alloc(IB * DMODEL * 2);
    h16_t* k16    = (h16_t*)alloc(TB * DMODEL * 2);
    h16_t* vT16   = (h16_t*)alloc(TB * DMODEL * 2);   // [b,n,d,j]
    h16_t* r16    = (h16_t*)alloc(TB * DMODEL * 2);
    float* vecf   = (float*)alloc(IB * DMODEL * 4);
    h16_t* vec16  = (h16_t*)alloc(IB * DMODEL * 2);
    float* aoutf  = (float*)alloc(IB * DMODEL * 4);
    float* hf     = (float*)alloc(IB * DMODEL * 4);
    h16_t* h16b   = (h16_t*)alloc(IB * DMODEL * 2);
    float* ffh    = (float*)alloc(IB * DFF * 4);
    h16_t* ffh16  = (h16_t*)alloc(IB * DFF * 2);
    float* ff2    = (float*)alloc(IB * DMODEL * 4);

    float* outp  = (float*)d_out;
    float* attnM = outp + IB * DMODEL;         // [XLEN, TOT]

    auto nb = [](size_t n) { return dim3((unsigned)((n + 255) / 256)); };

    // attn_matrix accumulates via atomics -> zero it every call
    k_zero_f32<<<nb((size_t)XLEN * TOT), 256, 0, stream>>>(attnM, XLEN * TOT);

    // fp32 -> fp16 staging
    k_cvt_f32_f16<<<nb((size_t)MEMLEN * BATCH * DMODEL), 256, 0, stream>>>(
        mem, c16, MEMLEN * BATCH * DMODEL);
    k_cvt_f32_f16<<<nb(IB * DMODEL), 256, 0, stream>>>(
        x, c16 + (size_t)MEMLEN * BATCH * DMODEL, (int)(IB * DMODEL));
    k_cvt_f32_f16<<<nb(IB * DMODEL), 256, 0, stream>>>(x, x16, (int)(IB * DMODEL));
    k_cvt_f32_f16<<<nb(TB * DMODEL), 256, 0, stream>>>(pos, pos16, (int)(TB * DMODEL));
    k_cvt_f32_f16<<<nb((size_t)DMODEL * DMODEL), 256, 0, stream>>>(Wq, Wq16, DMODEL * DMODEL);
    k_cvt_f32_f16<<<nb((size_t)DMODEL * 2 * DMODEL), 256, 0, stream>>>(Wkv, Wkv16, DMODEL * 2 * DMODEL);
    k_cvt_f32_f16<<<nb((size_t)DMODEL * DMODEL), 256, 0, stream>>>(Wrel, Wrel16, DMODEL * DMODEL);
    k_cvt_f32_f16<<<nb((size_t)DMODEL * DMODEL), 256, 0, stream>>>(Wo, Wo16, DMODEL * DMODEL);
    k_cvt_f32_f16<<<nb((size_t)DMODEL * DFF), 256, 0, stream>>>(W1, W116, DMODEL * DFF);
    k_cvt_f32_f16<<<nb((size_t)DFF * DMODEL), 256, 0, stream>>>(W2, W216, DFF * DMODEL);

    // projections (WMMA GEMMs): kv, q, r
    k_gemm_f16<<<dim3((2 * DMODEL) / 128, (unsigned)(TB / 128)), 256, 0, stream>>>(
        c16, Wkv16, kvf, nullptr, (int)TB, 2 * DMODEL, DMODEL, 0);
    k_gemm_f16<<<dim3(DMODEL / 128, (unsigned)(IB / 128)), 256, 0, stream>>>(
        x16, Wq16, qf, nullptr, (int)IB, DMODEL, DMODEL, 0);
    k_gemm_f16<<<dim3(DMODEL / 128, (unsigned)(TB / 128)), 256, 0, stream>>>(
        pos16, Wrel16, rf, nullptr, (int)TB, DMODEL, DMODEL, 0);

    // head-space prep
    k_quqv<<<nb(IB * DMODEL), 256, 0, stream>>>(qf, pbu, pbv, qu16, qv16, (int)(IB * DMODEL));
    k_splitk<<<nb(TB * DMODEL), 256, 0, stream>>>(kvf, k16, (int)(TB * DMODEL));
    k_transposeV<<<dim3(TOT / 64, BATCH * NHEAD), 256, 0, stream>>>(kvf, vT16);
    k_cvt_f32_f16<<<nb(TB * DMODEL), 256, 0, stream>>>(rf, r16, (int)(TB * DMODEL));

    // fused rel-attention (WMMA scores + WMMA prob@V)
    k_attention<<<dim3(XLEN / 64, BATCH * NHEAD), 128, 0, stream>>>(
        qu16, qv16, k16, vT16, r16, vecf, attnM);

    // output projection + residual + LN1
    k_cvt_f32_f16<<<nb(IB * DMODEL), 256, 0, stream>>>(vecf, vec16, (int)(IB * DMODEL));
    k_gemm_f16<<<dim3(DMODEL / 128, (unsigned)(IB / 128)), 256, 0, stream>>>(
        vec16, Wo16, aoutf, nullptr, (int)IB, DMODEL, DMODEL, 0);
    k_layernorm<<<(unsigned)IB, 256, 0, stream>>>(x, aoutf, ln1g, ln1b, hf);

    // FFN
    k_cvt_f32_f16<<<nb(IB * DMODEL), 256, 0, stream>>>(hf, h16b, (int)(IB * DMODEL));
    k_gemm_f16<<<dim3(DFF / 128, (unsigned)(IB / 128)), 256, 0, stream>>>(
        h16b, W116, ffh, b1, (int)IB, DFF, DMODEL, 1);
    k_cvt_f32_f16<<<nb(IB * DFF), 256, 0, stream>>>(ffh, ffh16, (int)(IB * DFF));
    k_gemm_f16<<<dim3(DMODEL / 128, (unsigned)(IB / 128)), 256, 0, stream>>>(
        ffh16, W216, ff2, b2, (int)IB, DMODEL, DFF, 0);

    // final residual + LN2 -> d_out
    k_layernorm<<<(unsigned)IB, 256, 0, stream>>>(hf, ff2, ln2g, ln2b, outp);
}
